// InteractionGNNCell_86088324481259
// MI455X (gfx1250) — matrix operand order repaired and checked by reading, not compiled, not run on hardware
//
#include <hip/hip_runtime.h>
#include <hip/hip_bf16.h>

#define D_      128
#define N_      50000
#define E_      625000
#define EPS_    1e-5f

typedef __attribute__((ext_vector_type(16))) _Float16 v16h;
typedef __attribute__((ext_vector_type(8)))  _Float16 v8h;
typedef __attribute__((ext_vector_type(8)))  float    v8f;

union V16H { v16h v; v8h h[2]; };

// ---------------------------------------------------------------------------
// Workspace layout (bytes, all 64B aligned)
// ---------------------------------------------------------------------------
#define WS_MSG     0                               // N*128 f32  = 25,600,000
#define WS_CS      (WS_MSG  + N_*128*4)            // N ints
#define WS_CD      (WS_CS   + N_*4)                // N ints
#define WS_NSC     (WS_CD   + N_*4)                // 256 f32 scale
#define WS_NSH     (WS_NSC  + 256*4)               // 256 f32 shift
#define WS_ESC     (WS_NSH  + 256*4)               // 384 f32 scale
#define WS_ESH     (WS_ESC  + 384*4)               // 384 f32 shift
#define WS_NW1     (WS_ESH  + 384*4)               // f16 [128][256]
#define WS_NW2     (WS_NW1  + 128*256*2)           // f16 [128][128]
#define WS_EW1     (WS_NW2  + 128*128*2)           // f16 [128][384]
#define WS_EW2     (WS_EW1  + 128*384*2)           // f16 [128][128]

// ---------------------------------------------------------------------------
// 0) zero message buffer + count arrays
// ---------------------------------------------------------------------------
__global__ void k_zero(float* msg, int* cnt) {
    int i = blockIdx.x * blockDim.x + threadIdx.x;
    if (i < N_ * 128) msg[i] = 0.0f;
    if (i < 2 * N_)   cnt[i] = 0;
}

// ---------------------------------------------------------------------------
// 1) convert + transpose weights to f16 col-major ([out][in])
// ---------------------------------------------------------------------------
__global__ void k_prep_weights(const float* nW1, const float* nW2,
                               const float* eW1, const float* eW2,
                               _Float16* nW1t, _Float16* nW2t,
                               _Float16* eW1t, _Float16* eW2t) {
    int tid = blockIdx.x * blockDim.x + threadIdx.x;
    const int n1 = 256 * 128, n2 = 128 * 128, n3 = 384 * 128;
    if (tid < n1) {
        int k = tid / 128, n = tid % 128;
        nW1t[n * 256 + k] = (_Float16)nW1[tid];
    } else if (tid < n1 + n2) {
        int t = tid - n1; int k = t / 128, n = t % 128;
        nW2t[n * 128 + k] = (_Float16)nW2[t];
    } else if (tid < n1 + n2 + n3) {
        int t = tid - n1 - n2; int k = t / 128, n = t % 128;
        eW1t[n * 384 + k] = (_Float16)eW1[t];
    } else if (tid < n1 + n2 + n3 + n2) {
        int t = tid - n1 - n2 - n3; int k = t / 128, n = t % 128;
        eW2t[n * 128 + k] = (_Float16)eW2[t];
    }
}

// ---------------------------------------------------------------------------
// 2) scatter-add edges into per-node messages + src/dst occurrence counts
//    (message buffer fits L2 -> atomics resolve at L2)
// ---------------------------------------------------------------------------
__global__ void k_scatter(const float* __restrict__ edges,
                          const int* __restrict__ graph,
                          float* msg, int* cs, int* cd) {
    int t = blockIdx.x * blockDim.x + threadIdx.x;   // E*32 threads
    if (t >= E_ * 32) return;
    int e   = t >> 5;
    int c4  = t & 31;
    int col = c4 * 4;
    int dst = graph[E_ + e];
    float4 v = *(const float4*)(edges + (long long)e * 128 + col);
    float* p = msg + (long long)dst * 128 + col;
    atomicAdd(p + 0, v.x); atomicAdd(p + 1, v.y);
    atomicAdd(p + 2, v.z); atomicAdd(p + 3, v.w);
    if (c4 == 0) {
        atomicAdd(cs + graph[e], 1);
        atomicAdd(cd + dst, 1);
    }
}

// ---------------------------------------------------------------------------
// block reduction helper (sum, sumsq)
// ---------------------------------------------------------------------------
__device__ __forceinline__ void block_reduce2(float& s, float& s2, float* sh) {
    int t = threadIdx.x;
    sh[t] = s; sh[256 + t] = s2;
    __syncthreads();
    for (int w = 128; w > 0; w >>= 1) {
        if (t < w) { sh[t] += sh[t + w]; sh[256 + t] += sh[256 + t + w]; }
        __syncthreads();
    }
    s = sh[0]; s2 = sh[256];
}

// ---------------------------------------------------------------------------
// 3) node BatchNorm stats: one block per column (256 blocks)
//    columns 0..127 from nodes, 128..255 from msg. y = x*scale + shift
// ---------------------------------------------------------------------------
__global__ __launch_bounds__(256)
void k_node_stats(const float* __restrict__ nodes, const float* __restrict__ msg,
                  const float* gamma, const float* beta,
                  float* scale, float* shift) {
    __shared__ float sh[512];
    int c = blockIdx.x;
    const float* base = (c < 128) ? (nodes + c) : (msg + (c - 128));
    float s = 0.f, s2 = 0.f;
    for (int r = threadIdx.x; r < N_; r += 256) {
        float x = base[(long long)r * 128];
        s += x; s2 += x * x;
    }
    block_reduce2(s, s2, sh);
    if (threadIdx.x == 0) {
        float mean = s / (float)N_;
        float var  = s2 / (float)N_ - mean * mean;
        float sc   = gamma[c] * rsqrtf(var + EPS_);
        scale[c] = sc;
        shift[c] = beta[c] - mean * sc;
    }
}

// ---------------------------------------------------------------------------
// 4) node MLP: BN -> f16 -> GEMM(256x128) -> GELU -> GEMM(128x128) -> +res
//    one block = one 16-row tile; 8 waves, each owns 16 output cols.
// ---------------------------------------------------------------------------
#define AS_N 264   // 256 + 8 pad halves
#define HS_  136   // 128 + 8 pad halves

__global__ __launch_bounds__(256)
void k_node_mlp(const float* __restrict__ nodes, const float* __restrict__ msg,
                const float* __restrict__ scale, const float* __restrict__ shift,
                const _Float16* __restrict__ W1t, const float* __restrict__ b1,
                const _Float16* __restrict__ W2t, const float* __restrict__ b2,
                float* __restrict__ out_nodes) {
    __shared__ _Float16 At[16 * AS_N];
    __shared__ _Float16 Ht[16 * HS_];
    const int row0 = blockIdx.x * 16;        // N = 50000 = 3125 * 16, no tail
    const int tid  = threadIdx.x;

    // stage BN(input) as f16 tile [16][256]
    {
        int r  = tid >> 4;
        int c0 = (tid & 15) * 16;
        long long gr = row0 + r;
        #pragma unroll
        for (int j = 0; j < 16; j++) {
            int c = c0 + j;
            float x = (c < 128) ? nodes[gr * 128 + c] : msg[gr * 128 + (c - 128)];
            At[r * AS_N + c] = (_Float16)(x * scale[c] + shift[c]);
        }
    }
    __syncthreads();

    const int lane = tid & 31;
    const int col0 = (tid >> 5) * 16;        // wave's 16-column strip
    const int n    = lane & 15;
    const int hi   = lane >> 4;
    const int m    = n;                      // A row held by this lane

    // ---- GEMM1: [16,256] x [256,128] ----
    v8f acc = {};
    #pragma unroll
    for (int kc = 0; kc < 8; kc++) {
        int kb = kc * 32;
        V16H a, b;
        a.h[0] = *(const v8h*)(At + m * AS_N + kb +      hi * 8);
        a.h[1] = *(const v8h*)(At + m * AS_N + kb + 16 + hi * 8);
        const _Float16* wp = W1t + (col0 + n) * 256 + kb + hi * 8;
        b.h[0] = *(const v8h*)(wp);
        b.h[1] = *(const v8h*)(wp + 16);
        acc = __builtin_amdgcn_wmma_f32_16x16x32_f16(false, a.v, false, b.v,
                                                     (short)0, acc, false, false);
    }
    // bias + exact GELU, stage h
    {
        float bv = b1[col0 + n];
        #pragma unroll
        for (int r = 0; r < 8; r++) {
            float x = acc[r] + bv;
            float g = 0.5f * x * (1.0f + erff(x * 0.70710678118654752f));
            Ht[(r + 8 * hi) * HS_ + col0 + n] = (_Float16)g;
        }
    }
    __syncthreads();

    // ---- GEMM2: [16,128] x [128,128] ----
    v8f acc2 = {};
    #pragma unroll
    for (int kc = 0; kc < 4; kc++) {
        int kb = kc * 32;
        V16H a, b;
        a.h[0] = *(const v8h*)(Ht + m * HS_ + kb +      hi * 8);
        a.h[1] = *(const v8h*)(Ht + m * HS_ + kb + 16 + hi * 8);
        const _Float16* wp = W2t + (col0 + n) * 128 + kb + hi * 8;
        b.h[0] = *(const v8h*)(wp);
        b.h[1] = *(const v8h*)(wp + 16);
        acc2 = __builtin_amdgcn_wmma_f32_16x16x32_f16(false, a.v, false, b.v,
                                                      (short)0, acc2, false, false);
    }
    float b2v = b2[col0 + n];
    #pragma unroll
    for (int r = 0; r < 8; r++) {
        long long gr = row0 + r + 8 * hi;
        out_nodes[gr * 128 + col0 + n] =
            acc2[r] + b2v + nodes[gr * 128 + col0 + n];
    }
}

// ---------------------------------------------------------------------------
// 5) edge BatchNorm stats (384 blocks):
//    cols 0..255 : count-weighted sums over updated nodes (avoids E-gather)
//    cols 256..383: direct column sums over edges
// ---------------------------------------------------------------------------
__global__ __launch_bounds__(256)
void k_edge_stats(const float* __restrict__ newnodes, const float* __restrict__ edges,
                  const int* __restrict__ cs, const int* __restrict__ cd,
                  const float* gamma, const float* beta,
                  float* scale, float* shift) {
    __shared__ float sh[512];
    int c = blockIdx.x;
    float s = 0.f, s2 = 0.f;
    if (c < 256) {
        const int* cnt = (c < 128) ? cs : cd;
        int col = c & 127;
        for (int r = threadIdx.x; r < N_; r += 256) {
            float x = newnodes[(long long)r * 128 + col];
            float w = (float)cnt[r];
            s += w * x; s2 += w * x * x;
        }
    } else {
        int col = c - 256;
        for (int r = threadIdx.x; r < E_; r += 256) {
            float x = edges[(long long)r * 128 + col];
            s += x; s2 += x * x;
        }
    }
    block_reduce2(s, s2, sh);
    if (threadIdx.x == 0) {
        float mean = s / (float)E_;
        float var  = s2 / (float)E_ - mean * mean;
        float sc   = gamma[c] * rsqrtf(var + EPS_);
        scale[c] = sc;
        shift[c] = beta[c] - mean * sc;
    }
}

// ---------------------------------------------------------------------------
// 6) edge MLP: gather -> BN -> GEMM(384x128) -> GELU -> GEMM(128x128) -> +res
// ---------------------------------------------------------------------------
#define AS_E 392   // 384 + 8 pad halves

__global__ __launch_bounds__(256)
void k_edge_mlp(const float* __restrict__ newnodes, const float* __restrict__ edges,
                const int* __restrict__ graph,
                const float* __restrict__ scale, const float* __restrict__ shift,
                const _Float16* __restrict__ W1t, const float* __restrict__ b1,
                const _Float16* __restrict__ W2t, const float* __restrict__ b2,
                float* __restrict__ out_edges) {
    __shared__ _Float16 At[16 * AS_E];
    __shared__ _Float16 Ht[16 * HS_];
    const int row0 = blockIdx.x * 16;
    const int tid  = threadIdx.x;

    // stage BN([nodes[src], nodes[dst], edges]) as f16 tile [16][384]
    {
        int r  = tid >> 4;
        int c0 = (tid & 15) * 24;
        int eg = row0 + r;
        bool valid = eg < E_;
        int si = valid ? graph[eg]      : 0;
        int di = valid ? graph[E_ + eg] : 0;
        #pragma unroll
        for (int j = 0; j < 24; j++) {
            int c = c0 + j;
            float x;
            if (!valid)       x = 0.0f;
            else if (c < 128) x = newnodes[(long long)si * 128 + c];
            else if (c < 256) x = newnodes[(long long)di * 128 + (c - 128)];
            else              x = edges[(long long)eg * 128 + (c - 256)];
            At[r * AS_E + c] = (_Float16)(x * scale[c] + shift[c]);
        }
    }
    __syncthreads();

    const int lane = tid & 31;
    const int col0 = (tid >> 5) * 16;
    const int n    = lane & 15;
    const int hi   = lane >> 4;
    const int m    = n;

    // ---- GEMM1: [16,384] x [384,128] ----
    v8f acc = {};
    #pragma unroll
    for (int kc = 0; kc < 12; kc++) {
        int kb = kc * 32;
        V16H a, b;
        a.h[0] = *(const v8h*)(At + m * AS_E + kb +      hi * 8);
        a.h[1] = *(const v8h*)(At + m * AS_E + kb + 16 + hi * 8);
        const _Float16* wp = W1t + (col0 + n) * 384 + kb + hi * 8;
        b.h[0] = *(const v8h*)(wp);
        b.h[1] = *(const v8h*)(wp + 16);
        acc = __builtin_amdgcn_wmma_f32_16x16x32_f16(false, a.v, false, b.v,
                                                     (short)0, acc, false, false);
    }
    {
        float bv = b1[col0 + n];
        #pragma unroll
        for (int r = 0; r < 8; r++) {
            float x = acc[r] + bv;
            float g = 0.5f * x * (1.0f + erff(x * 0.70710678118654752f));
            Ht[(r + 8 * hi) * HS_ + col0 + n] = (_Float16)g;
        }
    }
    __syncthreads();

    // ---- GEMM2: [16,128] x [128,128] ----
    v8f acc2 = {};
    #pragma unroll
    for (int kc = 0; kc < 4; kc++) {
        int kb = kc * 32;
        V16H a, b;
        a.h[0] = *(const v8h*)(Ht + m * HS_ + kb +      hi * 8);
        a.h[1] = *(const v8h*)(Ht + m * HS_ + kb + 16 + hi * 8);
        const _Float16* wp = W2t + (col0 + n) * 128 + kb + hi * 8;
        b.h[0] = *(const v8h*)(wp);
        b.h[1] = *(const v8h*)(wp + 16);
        acc2 = __builtin_amdgcn_wmma_f32_16x16x32_f16(false, a.v, false, b.v,
                                                      (short)0, acc2, false, false);
    }
    float b2v = b2[col0 + n];
    #pragma unroll
    for (int r = 0; r < 8; r++) {
        long long eg = row0 + r + 8 * hi;
        if (eg < E_)
            out_edges[eg * 128 + col0 + n] =
                acc2[r] + b2v + edges[eg * 128 + col0 + n];
    }
}

// ---------------------------------------------------------------------------
// launch
// ---------------------------------------------------------------------------
extern "C" void kernel_launch(void* const* d_in, const int* in_sizes, int n_in,
                              void* d_out, int out_size, void* d_ws, size_t ws_size,
                              hipStream_t stream) {
    const float* nodes  = (const float*)d_in[0];
    const float* edges  = (const float*)d_in[1];
    const int*   graph  = (const int*)  d_in[2];
    const float* ngamma = (const float*)d_in[3];
    const float* nbeta  = (const float*)d_in[4];
    const float* egamma = (const float*)d_in[5];
    const float* ebeta  = (const float*)d_in[6];
    const float* nW1    = (const float*)d_in[7];
    const float* nb1    = (const float*)d_in[8];
    const float* nW2    = (const float*)d_in[9];
    const float* nb2    = (const float*)d_in[10];
    const float* eW1    = (const float*)d_in[11];
    const float* eb1    = (const float*)d_in[12];
    const float* eW2    = (const float*)d_in[13];
    const float* eb2    = (const float*)d_in[14];

    char* ws = (char*)d_ws;
    float*     msg   = (float*)    (ws + WS_MSG);
    int*       cs    = (int*)      (ws + WS_CS);
    int*       cd    = (int*)      (ws + WS_CD);
    float*     nsc   = (float*)    (ws + WS_NSC);
    float*     nsh   = (float*)    (ws + WS_NSH);
    float*     esc   = (float*)    (ws + WS_ESC);
    float*     esh   = (float*)    (ws + WS_ESH);
    _Float16*  nW1t  = (_Float16*) (ws + WS_NW1);
    _Float16*  nW2t  = (_Float16*) (ws + WS_NW2);
    _Float16*  eW1t  = (_Float16*) (ws + WS_EW1);
    _Float16*  eW2t  = (_Float16*) (ws + WS_EW2);

    float* out_nodes = (float*)d_out;
    float* out_edges = out_nodes + (long long)N_ * 128;

    // 0) zero msg + counts
    k_zero<<<(N_ * 128 + 255) / 256, 256, 0, stream>>>(msg, cs);

    // 1) weight convert + transpose (114688 elems)
    k_prep_weights<<<448, 256, 0, stream>>>(nW1, nW2, eW1, eW2,
                                            nW1t, nW2t, eW1t, eW2t);

    // 2) scatter-add messages + counts (E*32 threads)
    k_scatter<<<E_ * 32 / 256, 256, 0, stream>>>(edges, graph, msg, cs, cd);

    // 3) node BN stats
    k_node_stats<<<256, 256, 0, stream>>>(nodes, msg, ngamma, nbeta, nsc, nsh);

    // 4) node MLP (3125 row tiles)
    k_node_mlp<<<N_ / 16, 256, 0, stream>>>(nodes, msg, nsc, nsh,
                                            nW1t, nb1, nW2t, nb2, out_nodes);

    // 5) edge BN stats
    k_edge_stats<<<384, 256, 0, stream>>>(out_nodes, edges, cs, cd,
                                          egamma, ebeta, esc, esh);

    // 6) edge MLP (39063 row tiles)
    k_edge_mlp<<<(E_ + 15) / 16, 256, 0, stream>>>(out_nodes, edges, graph,
                                                   esc, esh, eW1t, eb1,
                                                   eW2t, eb2, out_edges);
}
